// PointNet2SSGSemSeg_4724464025876
// MI455X (gfx1250) — matrix operand order repaired and checked by reading, not compiled, not run on hardware
//
#include <hip/hip_runtime.h>

typedef _Float16 h8   __attribute__((ext_vector_type(8)));
typedef _Float16 v16h __attribute__((ext_vector_type(16)));
typedef float    v8f  __attribute__((ext_vector_type(8)));

static inline int cdiv(int a, int b) { return (a + b - 1) / b; }
static inline int rup32(int x) { return (x + 31) & ~31; }
static inline int rup64(int x) { return (x + 63) & ~63; }

// ---------------------------------------------------------------------------
// points (B,6,N) -> xyz (B,N,3), feat (B,N,3)
__global__ void split_k(const float* __restrict__ pts, float* __restrict__ xyz,
                        float* __restrict__ feat, int B, int N) {
    int i = blockIdx.x * blockDim.x + threadIdx.x;
    if (i >= B * N) return;
    int b = i / N, p = i % N;
    const float* s = pts + (size_t)b * 6 * N + p;
    xyz[(size_t)i * 3 + 0] = s[0 * N];
    xyz[(size_t)i * 3 + 1] = s[1 * N];
    xyz[(size_t)i * 3 + 2] = s[2 * N];
    feat[(size_t)i * 3 + 0] = s[3 * N];
    feat[(size_t)i * 3 + 1] = s[4 * N];
    feat[(size_t)i * 3 + 2] = s[5 * N];
}

// ---------------------------------------------------------------------------
// Farthest point sampling: one block per batch, dists live in LDS.
__global__ __launch_bounds__(256) void fps_k(const float* __restrict__ xyz,
                                             int* __restrict__ out, int n, int m) {
    __shared__ float dist[8192];
    __shared__ float rval[256];
    __shared__ int   rarg[256];
    __shared__ int   s_far;
    int b = blockIdx.x, t = threadIdx.x;
    const float* P = xyz + (size_t)b * n * 3;
    for (int i = t; i < n; i += 256) dist[i] = 1e10f;
    if (t == 0) s_far = 0;
    __syncthreads();
    for (int it = 0; it < m; ++it) {
        int far = s_far;
        if (t == 0) out[b * m + it] = far;
        float cx = P[far * 3 + 0], cy = P[far * 3 + 1], cz = P[far * 3 + 2];
        float best = -1.0f; int barg = 0;
        for (int i = t; i < n; i += 256) {
            float dx = P[i * 3 + 0] - cx, dy = P[i * 3 + 1] - cy, dz = P[i * 3 + 2] - cz;
            float d = dx * dx + dy * dy + dz * dz;
            float dm = fminf(dist[i], d);
            dist[i] = dm;
            if (dm > best) { best = dm; barg = i; }
        }
        rval[t] = best; rarg[t] = barg;
        __syncthreads();
        for (int s = 128; s > 0; s >>= 1) {
            if (t < s) {
                if (rval[t + s] > rval[t] ||
                    (rval[t + s] == rval[t] && rarg[t + s] < rarg[t])) {
                    rval[t] = rval[t + s]; rarg[t] = rarg[t + s];
                }
            }
            __syncthreads();
        }
        if (t == 0) s_far = rarg[0];
        __syncthreads();
    }
}

__global__ void gather_xyz_k(const float* __restrict__ xyz, const int* __restrict__ idx,
                             float* __restrict__ out, int total, int n, int M) {
    int i = blockIdx.x * blockDim.x + threadIdx.x;
    if (i >= total) return;
    int b = i / M;
    int s = idx[i];
    const float* p = xyz + ((size_t)b * n + s) * 3;
    out[(size_t)i * 3 + 0] = p[0];
    out[(size_t)i * 3 + 1] = p[1];
    out[(size_t)i * 3 + 2] = p[2];
}

// Ball query: first K indices (ascending) with d2 < r2, padded with first hit.
__global__ void ballq_k(const float* __restrict__ nxyz, const float* __restrict__ xyz,
                        int* __restrict__ nidx, int total, int M, int n, float r2, int K) {
    int i = blockIdx.x * blockDim.x + threadIdx.x;
    if (i >= total) return;
    int b = i / M;
    const float* q = nxyz + (size_t)i * 3;
    const float* P = xyz + (size_t)b * n * 3;
    int* o = nidx + (size_t)i * K;
    float qx = q[0], qy = q[1], qz = q[2];
    int cnt = 0, first = 0;
    for (int p = 0; p < n && cnt < K; ++p) {
        float dx = P[p * 3 + 0] - qx, dy = P[p * 3 + 1] - qy, dz = P[p * 3 + 2] - qz;
        if (dx * dx + dy * dy + dz * dz < r2) {
            if (cnt == 0) first = p;
            o[cnt++] = p;
        }
    }
    for (int j = cnt; j < K; ++j) o[j] = first;
}

// Build grouped GEMM input rows: [rel_xyz(3) | feat(Cf) | zeropad] as f16.
__global__ void group_pack_k(const float* __restrict__ xyz, const float* __restrict__ feat,
                             const float* __restrict__ nxyz, const int* __restrict__ nidx,
                             _Float16* __restrict__ X, int total, int n, int M, int K,
                             int Cf, int Cpad) {
    int i = blockIdx.x * blockDim.x + threadIdx.x;
    if (i >= total) return;
    int b = i / (M * K);
    int mm = (i % (M * K)) / K;
    int src = nidx[i];
    const float* p = xyz + ((size_t)b * n + src) * 3;
    const float* q = nxyz + ((size_t)b * M + mm) * 3;
    _Float16* x = X + (size_t)i * Cpad;
    x[0] = (_Float16)(p[0] - q[0]);
    x[1] = (_Float16)(p[1] - q[1]);
    x[2] = (_Float16)(p[2] - q[2]);
    const float* f = feat + ((size_t)b * n + src) * Cf;
    for (int c = 0; c < Cf; ++c) x[3 + c] = (_Float16)f[c];
    for (int c = 3 + Cf; c < Cpad; ++c) x[c] = (_Float16)0.f;
}

// Pad+convert weights (O,C) f32 -> (Opad64,Cpad) f16 (zero rows for padding).
__global__ void wpack_k(const float* __restrict__ W, _Float16* __restrict__ W16,
                        int O, int C, int Opad, int Cpad) {
    int i = blockIdx.x * blockDim.x + threadIdx.x;
    if (i >= Opad * Cpad) return;
    int o = i / Cpad, c = i % Cpad;
    W16[i] = (o < O && c < C) ? (_Float16)W[(size_t)o * C + c] : (_Float16)0.f;
}

// ---------------------------------------------------------------------------
// WMMA GEMM: Y[r,o] = act(sum_c X[r,c]*W[o,c] + b[o]).
// One wave computes a 16x64 output tile: 4 accumulators share one A fragment,
// so per 32-wide K step each lane issues 2 A-loads + 8 B-loads for 4 WMMAs.
// A layout (16x32 f16): lane m=lane&15, half=lane>>4; e -> K=half*16+e.
// B layout (32x16 f16 from W rows): lane n=lane&15; e -> K=half*16+e.
// C/D f32: m=(lane>>4)*8+j, n=lane&15.
__global__ __launch_bounds__(256)
void gemm_wmma_k(const _Float16* __restrict__ X, const _Float16* __restrict__ W,
                 const float* __restrict__ bias, _Float16* __restrict__ Y16,
                 float* __restrict__ Yf32, int R, int Cpad, int O, int Opad, int relu) {
    int wave = (blockIdx.x * blockDim.x + threadIdx.x) >> 5;
    int lane = threadIdx.x & 31;
    int tg_count = Opad >> 6;                  // groups of 4 N-tiles
    int total = (R >> 4) * tg_count;
    if (wave >= total) return;
    int tm = wave / tg_count, tg = wave % tg_count;
    int row0 = tm << 4, oc0 = tg << 6;
    int lmod = lane & 15, half = lane >> 4;

    const _Float16* xrow = X + (size_t)(row0 + lmod) * Cpad;
    const _Float16* wr0 = W + (size_t)(oc0 + 0  + lmod) * Cpad;
    const _Float16* wr1 = W + (size_t)(oc0 + 16 + lmod) * Cpad;
    const _Float16* wr2 = W + (size_t)(oc0 + 32 + lmod) * Cpad;
    const _Float16* wr3 = W + (size_t)(oc0 + 48 + lmod) * Cpad;

    v8f acc0 = {0.f,0.f,0.f,0.f,0.f,0.f,0.f,0.f};
    v8f acc1 = acc0, acc2 = acc0, acc3 = acc0;
    for (int kb = 0; kb < Cpad; kb += 32) {
        union { v16h v; h8 h[2]; } a, b0, b1, b2, b3;
        a.h[0]  = *(const h8*)(xrow + kb + half * 16);
        a.h[1]  = *(const h8*)(xrow + kb + half * 16 + 8);
        b0.h[0] = *(const h8*)(wr0 + kb + half * 16);
        b0.h[1] = *(const h8*)(wr0 + kb + half * 16 + 8);
        b1.h[0] = *(const h8*)(wr1 + kb + half * 16);
        b1.h[1] = *(const h8*)(wr1 + kb + half * 16 + 8);
        b2.h[0] = *(const h8*)(wr2 + kb + half * 16);
        b2.h[1] = *(const h8*)(wr2 + kb + half * 16 + 8);
        b3.h[0] = *(const h8*)(wr3 + kb + half * 16);
        b3.h[1] = *(const h8*)(wr3 + kb + half * 16 + 8);
        acc0 = __builtin_amdgcn_wmma_f32_16x16x32_f16(false, a.v, false, b0.v,
                                                      (short)0, acc0, false, false);
        acc1 = __builtin_amdgcn_wmma_f32_16x16x32_f16(false, a.v, false, b1.v,
                                                      (short)0, acc1, false, false);
        acc2 = __builtin_amdgcn_wmma_f32_16x16x32_f16(false, a.v, false, b2.v,
                                                      (short)0, acc2, false, false);
        acc3 = __builtin_amdgcn_wmma_f32_16x16x32_f16(false, a.v, false, b3.v,
                                                      (short)0, acc3, false, false);
    }
    int mbase = row0 + half * 8;
#pragma unroll
    for (int t = 0; t < 4; ++t) {
        v8f acc = (t == 0) ? acc0 : (t == 1) ? acc1 : (t == 2) ? acc2 : acc3;
        int oc = oc0 + t * 16 + lmod;
        if (oc < O) {
            float bv = bias[oc];
#pragma unroll
            for (int j = 0; j < 8; ++j) {
                float v = acc[j] + bv;
                if (relu && v < 0.f) v = 0.f;
                int m = mbase + j;
                if (Yf32) Yf32[(size_t)m * O + oc] = v;
                if (Y16)  Y16[(size_t)m * O + oc] = (_Float16)v;
            }
        }
    }
}

// Max over K neighbours: Y [BM,K,C] f32 -> F [BM,C] f32.
__global__ void maxpool_k(const float* __restrict__ Y, float* __restrict__ F,
                          int BM, int K, int C) {
    int i = blockIdx.x * blockDim.x + threadIdx.x;
    if (i >= BM * C) return;
    int bm = i / C, c = i % C;
    const float* p = Y + (size_t)bm * K * C + c;
    float m = p[0];
    for (int k = 1; k < K; ++k) m = fmaxf(m, p[(size_t)k * C]);
    F[(size_t)bm * C + c] = m;
}

// concat [xyz(3)|feat(Cf)] f32 per point
__global__ void concat_k(const float* __restrict__ xyz, const float* __restrict__ feat,
                         float* __restrict__ out, int BM, int Cf) {
    int C = 3 + Cf;
    int i = blockIdx.x * blockDim.x + threadIdx.x;
    if (i >= BM * C) return;
    int p = i / C, c = i % C;
    out[i] = (c < 3) ? xyz[(size_t)p * 3 + c] : feat[(size_t)p * Cf + (c - 3)];
}

// FP module front-end: 3-NN inverse-distance interp + concat -> f16 GEMM input.
__global__ void fp_pack_k(const float* __restrict__ sxyz, const float* __restrict__ dxyz,
                          const float* __restrict__ sfeat, const float* __restrict__ dfeat,
                          _Float16* __restrict__ X, int total, int S, int D,
                          int Sc, int Dc, int Cpad) {
    int i = blockIdx.x * blockDim.x + threadIdx.x;
    if (i >= total) return;
    int b = i / S;
    const float* q = sxyz + (size_t)i * 3;
    const float* P = dxyz + (size_t)b * D * 3;
    float d0 = 3.4e38f, d1 = 3.4e38f, d2 = 3.4e38f;
    int i0 = 0, i1 = 0, i2 = 0;
    for (int p = 0; p < D; ++p) {
        float dx = P[p * 3 + 0] - q[0], dy = P[p * 3 + 1] - q[1], dz = P[p * 3 + 2] - q[2];
        float d = dx * dx + dy * dy + dz * dz;
        if (d < d0)      { d2 = d1; i2 = i1; d1 = d0; i1 = i0; d0 = d; i0 = p; }
        else if (d < d1) { d2 = d1; i2 = i1; d1 = d; i1 = p; }
        else if (d < d2) { d2 = d; i2 = p; }
    }
    float w0 = 1.f / (d0 + 1e-8f), w1 = 1.f / (d1 + 1e-8f), w2 = 1.f / (d2 + 1e-8f);
    float ws = w0 + w1 + w2;
    w0 /= ws; w1 /= ws; w2 /= ws;
    const float* F = dfeat + (size_t)b * D * Dc;
    _Float16* x = X + (size_t)i * Cpad;
    for (int c = 0; c < Dc; ++c)
        x[c] = (_Float16)(w0 * F[(size_t)i0 * Dc + c] + w1 * F[(size_t)i1 * Dc + c] +
                          w2 * F[(size_t)i2 * Dc + c]);
    const float* sf = sfeat + (size_t)i * Sc;
    for (int c = 0; c < Sc; ++c) x[Dc + c] = (_Float16)sf[c];
    for (int c = Dc + Sc; c < Cpad; ++c) x[c] = (_Float16)0.f;
}

// head staging [B*M,O] -> d_out (B,O,M)
__global__ void transpose_out_k(const float* __restrict__ Yin, float* __restrict__ out,
                                int B, int M, int O) {
    int i = blockIdx.x * blockDim.x + threadIdx.x;
    if (i >= B * M * O) return;
    int b = i / (M * O);
    int r = i % (M * O);
    int mm = r / O, o = r % O;
    out[((size_t)b * O + o) * M + mm] = Yin[((size_t)b * M + mm) * O + o];
}

// ---------------------------------------------------------------------------
extern "C" void kernel_launch(void* const* d_in, const int* in_sizes, int n_in,
                              void* d_out, int out_size, void* d_ws, size_t ws_size,
                              hipStream_t stream) {
    (void)out_size; (void)ws_size;
    const int B = 4, N = 8192, K = 32;
    const int Ms[4]   = {1024, 256, 64, 16};
    const float rad[4] = {0.1f, 0.2f, 0.4f, 0.8f};
    const int saO[4][3] = {{32,32,64},{64,64,128},{128,128,256},{256,256,512}};
    const int saCin[4]  = {6, 67, 131, 259};
    const int fpO[4][3] = {{256,256,0},{256,256,0},{256,128,0},{128,128,128}};
    const int fpCin[4]  = {1027, 512, 384, 192};
    const int fpL[4]    = {2, 2, 2, 3};

    // ---- locate inputs (hedge both flatten orders) ----
    const float* points;
    int pb;
    if (in_sizes[0] == B * 6 * N) { points = (const float*)d_in[0]; pb = 1; }
    else                         { points = (const float*)d_in[n_in - 1]; pb = 0; }
    // canonical leaves: sa(24: sa0 w0,b0,w1,b1,w2,b2 ...), fp(18), seg(2), head(2)
    const float* leaf[46];
    if (in_sizes[pb] == 32 * 6) {   // insertion order: sa, fp, seg, head
        for (int i = 0; i < 46; ++i) leaf[i] = (const float*)d_in[pb + i];
    } else {                        // jax sorted keys: fp, head, sa, seg
        for (int i = 0; i < 24; ++i) leaf[i]      = (const float*)d_in[pb + 20 + i];
        for (int i = 0; i < 18; ++i) leaf[24 + i] = (const float*)d_in[pb + i];
        leaf[42] = (const float*)d_in[pb + 44];  // seg_w
        leaf[43] = (const float*)d_in[pb + 45];  // seg_b
        leaf[44] = (const float*)d_in[pb + 18];  // head_w
        leaf[45] = (const float*)d_in[pb + 19];  // head_b
    }

    // ---- workspace bump allocator ----
    char* base = (char*)d_ws;
    size_t off = 0;
    auto alloc = [&](size_t bytes) -> void* {
        void* p = base + off;
        off = (off + bytes + 255) & ~(size_t)255;
        return p;
    };

    float* xyz0  = (float*)alloc((size_t)B * N * 3 * 4);
    float* feat0 = (float*)alloc((size_t)B * N * 3 * 4);
    float* xyz_lv[5]; float* feat_lv[5];
    xyz_lv[0] = xyz0; feat_lv[0] = feat0;
    for (int l = 0; l < 4; ++l) {
        xyz_lv[l + 1]  = (float*)alloc((size_t)B * Ms[l] * 3 * 4);
        feat_lv[l + 1] = (float*)alloc((size_t)B * Ms[l] * saO[l][2] * 4);
    }
    int*       fpsidx = (int*)alloc((size_t)B * 1024 * 4);
    int*       nidx   = (int*)alloc((size_t)B * 1024 * K * 4);
    _Float16*  X16    = (_Float16*)alloc((size_t)131072 * 32 * 2);   // max pack buffer
    _Float16*  Y16A   = (_Float16*)alloc((size_t)131072 * 32 * 2);
    _Float16*  Y16B   = (_Float16*)alloc((size_t)131072 * 32 * 2);
    float*     YF32   = (float*)alloc((size_t)131072 * 64 * 4);
    float*     dfeat0 = (float*)alloc((size_t)B * 16 * 515 * 4);
    float*     fpout[3];
    fpout[0] = (float*)alloc((size_t)B * 16 * 256 * 4);
    fpout[1] = (float*)alloc((size_t)B * 64 * 256 * 4);
    fpout[2] = (float*)alloc((size_t)B * 256 * 128 * 4);
    float*     headst = (float*)alloc((size_t)B * 1024 * 13 * 4);

    // ---- pack all weights to padded f16 (rows padded to 64) ----
    auto packW = [&](const float* W, int O, int C) -> _Float16* {
        int Op = rup64(O), Cp = rup32(C);
        _Float16* w16 = (_Float16*)alloc((size_t)Op * Cp * 2);
        wpack_k<<<cdiv(Op * Cp, 256), 256, 0, stream>>>(W, w16, O, C, Op, Cp);
        return w16;
    };
    _Float16* saW[4][3]; const float* saB[4][3];
    for (int i = 0; i < 4; ++i) {
        int c = saCin[i];
        for (int l = 0; l < 3; ++l) {
            saW[i][l] = packW(leaf[(i * 3 + l) * 2], saO[i][l], c);
            saB[i][l] = leaf[(i * 3 + l) * 2 + 1];
            c = saO[i][l];
        }
    }
    _Float16* fpW[4][3]; const float* fpB[4][3];
    {
        int ix = 24;
        for (int i = 0; i < 4; ++i) {
            int c = fpCin[i];
            for (int l = 0; l < fpL[i]; ++l) {
                fpW[i][l] = packW(leaf[ix], fpO[i][l], c);
                fpB[i][l] = leaf[ix + 1];
                ix += 2;
                c = fpO[i][l];
            }
        }
    }
    _Float16* segW = packW(leaf[42], 128, 128); const float* segB = leaf[43];
    _Float16* hdW  = packW(leaf[44], 13, 128);  const float* hdB  = leaf[45];

    auto gemm = [&](const _Float16* X, const _Float16* W, const float* bias,
                    _Float16* Y16, float* Yf32, int R, int C, int O, int relu) {
        int Cp = rup32(C), Op = rup64(O);
        int waves = (R / 16) * (Op / 64);
        gemm_wmma_k<<<cdiv(waves, 8), 256, 0, stream>>>(X, W, bias, Y16, Yf32,
                                                        R, Cp, O, Op, relu);
    };

    // ---- input split ----
    split_k<<<cdiv(B * N, 256), 256, 0, stream>>>(points, xyz0, feat0, B, N);

    // ---- SA levels ----
    int n = N, curCf = 3;
    for (int l = 0; l < 4; ++l) {
        int M = Ms[l];
        fps_k<<<B, 256, 0, stream>>>(xyz_lv[l], fpsidx, n, M);
        gather_xyz_k<<<cdiv(B * M, 256), 256, 0, stream>>>(xyz_lv[l], fpsidx,
                                                           xyz_lv[l + 1], B * M, n, M);
        ballq_k<<<cdiv(B * M, 256), 256, 0, stream>>>(xyz_lv[l + 1], xyz_lv[l], nidx,
                                                      B * M, M, n, rad[l] * rad[l], K);
        int Cin = 3 + curCf, Cpad = rup32(Cin);
        int rows = B * M * K;
        group_pack_k<<<cdiv(rows, 256), 256, 0, stream>>>(xyz_lv[l], feat_lv[l],
                                                          xyz_lv[l + 1], nidx, X16,
                                                          rows, n, M, K, curCf, Cpad);
        gemm(X16,  saW[l][0], saB[l][0], Y16A, nullptr, rows, Cin,      saO[l][0], 1);
        gemm(Y16A, saW[l][1], saB[l][1], Y16B, nullptr, rows, saO[l][0], saO[l][1], 1);
        gemm(Y16B, saW[l][2], saB[l][2], nullptr, YF32, rows, saO[l][1], saO[l][2], 1);
        maxpool_k<<<cdiv(B * M * saO[l][2], 256), 256, 0, stream>>>(YF32, feat_lv[l + 1],
                                                                    B * M, K, saO[l][2]);
        n = M;
        curCf = saO[l][2];
    }

    // ---- FP levels ----
    concat_k<<<cdiv(B * 16 * 515, 256), 256, 0, stream>>>(xyz_lv[4], feat_lv[4],
                                                          dfeat0, B * 16, 512);
    struct FPD { float* sxyz; float* dxyz; float* sfeat; float* dfeat;
                 int S, D, Sc, Dc; };
    FPD fpd[4] = {
        { xyz_lv[4], xyz_lv[4], feat_lv[4], dfeat0,   16,  16, 512, 515 },
        { xyz_lv[3], xyz_lv[4], feat_lv[3], fpout[0], 64,  16, 256, 256 },
        { xyz_lv[2], xyz_lv[3], feat_lv[2], fpout[1], 256, 64, 128, 256 },
        { xyz_lv[1], xyz_lv[2], feat_lv[1], fpout[2], 1024, 256, 64, 128 },
    };
    for (int i = 0; i < 4; ++i) {
        const FPD& f = fpd[i];
        int rows = B * f.S;
        int Cin = f.Dc + f.Sc, Cpad = rup32(Cin);
        fp_pack_k<<<cdiv(rows, 256), 256, 0, stream>>>(f.sxyz, f.dxyz, f.sfeat, f.dfeat,
                                                       X16, rows, f.S, f.D, f.Sc, f.Dc,
                                                       Cpad);
        if (fpL[i] == 2) {
            gemm(X16,  fpW[i][0], fpB[i][0], Y16A, nullptr, rows, Cin,      fpO[i][0], 1);
            gemm(Y16A, fpW[i][1], fpB[i][1], nullptr, fpout[i], rows, fpO[i][0], fpO[i][1], 1);
        } else { // fp3: keep f16 for seg/head
            gemm(X16,  fpW[i][0], fpB[i][0], Y16A, nullptr, rows, Cin,      fpO[i][0], 1);
            gemm(Y16A, fpW[i][1], fpB[i][1], Y16B, nullptr, rows, fpO[i][0], fpO[i][1], 1);
            gemm(Y16B, fpW[i][2], fpB[i][2], Y16A, nullptr, rows, fpO[i][1], fpO[i][2], 1);
        }
    }

    // ---- seg MLP + head ----
    int rows = B * 1024;
    gemm(Y16A, segW, segB, Y16B, nullptr, rows, 128, 128, 1);
    gemm(Y16B, hdW,  hdB,  nullptr, headst, rows, 128, 13, 0);
    transpose_out_k<<<cdiv(rows * 13, 256), 256, 0, stream>>>(headst, (float*)d_out,
                                                              B, 1024, 13);
}
